// BottleneckSparse2D_13331578487456
// MI455X (gfx1250) — matrix operand order repaired and checked by reading, not compiled, not run on hardware
//
#include <hip/hip_runtime.h>
#include <hip/hip_bf16.h>

typedef __attribute__((ext_vector_type(16))) __bf16 v16bf;
typedef __attribute__((ext_vector_type(8)))  __bf16 v8bf;
typedef __attribute__((ext_vector_type(8)))  float  v8f;
typedef __attribute__((ext_vector_type(8)))  int    v8i;

union BV16 { v16bf v; v8bf h[2]; v8i i; };

static __device__ __forceinline__ unsigned short f2bf(float f) {
    unsigned int u = __float_as_uint(f);
    unsigned int r = u + 0x7FFFu + ((u >> 16) & 1u);
    return (unsigned short)(r >> 16);
}

// ---------------------------------------------------------------------------
// Convert f32 -> bf16 (round to nearest even)
// ---------------------------------------------------------------------------
__global__ void k_f32_to_bf16(const float* __restrict__ x,
                              unsigned short* __restrict__ y, unsigned n) {
    unsigned stride = gridDim.x * blockDim.x;
    for (unsigned i = blockIdx.x * blockDim.x + threadIdx.x; i < n; i += stride)
        y[i] = f2bf(x[i]);
}

// ---------------------------------------------------------------------------
// Pack weight matrix (taps x K x Cout, f32) into WMMA B-operand layout (bf16):
// index = ((((tap*ktc + kt)*ntc + nt)*32 + lane)*16 + elem)
// lane group (lane>=16) covers K offset +8; elems 8..15 cover K offset +16.
// ---------------------------------------------------------------------------
__global__ void k_pack_w(const float* __restrict__ W, unsigned short* __restrict__ P,
                         int Cout, int ktc, int ntc, int taps) {
    unsigned total = (unsigned)taps * ktc * ntc * 512u;
    int K = ktc * 32;
    unsigned stride = gridDim.x * blockDim.x;
    for (unsigned i = blockIdx.x * blockDim.x + threadIdx.x; i < total; i += stride) {
        int elem = (int)(i & 15u);
        int lane = (int)((i >> 4) & 31u);
        unsigned t2 = i >> 9;                  // (tap*ktc + kt)*ntc + nt
        int nt = (int)(t2 % ntc);
        unsigned t3 = t2 / ntc;
        int kt = (int)(t3 % ktc);
        int tap = (int)(t3 / ktc);
        int k = kt * 32 + (elem >> 3) * 16 + (lane >> 4) * 8 + (elem & 7);
        int n = nt * 16 + (lane & 15);
        P[i] = f2bf(W[(unsigned)tap * K * Cout + (unsigned)k * Cout + n]);
    }
}

// ---------------------------------------------------------------------------
// Dense GEMM: C[N x COUT] = A[N x 64](bf16) * B(packed bf16), f32 accumulate.
// One wave32 per 16-row tile, 8 waves per block. N must be a multiple of 16.
// A loaded once into registers; per-lane addressing is 32-bit offsets off
// uniform SGPR bases; epilogue stores use immediate offsets (COUT constexpr).
// ---------------------------------------------------------------------------
template<int COUT, int NTC>
__global__ void k_gemm_bf16(const unsigned short* __restrict__ A,
                            const unsigned short* __restrict__ Bp,
                            float* __restrict__ C, int N) {
    int lane = (int)(threadIdx.x & 31u);
    int wave = (int)(threadIdx.x >> 5);
    int rowbase = (blockIdx.x * (int)(blockDim.x >> 5) + wave) * 16;
    if (rowbase >= N) return;                       // wave-uniform: EXEC stays full
    int lo = lane & 15, hi = lane >> 4;
    unsigned aoff = (unsigned)(rowbase + lo) * 64u + (unsigned)hi * 8u;
    unsigned boff0 = (unsigned)lane * 16u;
    unsigned coff = (unsigned)(rowbase + hi * 8) * (unsigned)COUT + (unsigned)lo;

    BV16 a0, a1;
    a0.h[0] = *reinterpret_cast<const v8bf*>(A + aoff);
    a0.h[1] = *reinterpret_cast<const v8bf*>(A + aoff + 16);
    a1.h[0] = *reinterpret_cast<const v8bf*>(A + aoff + 32);
    a1.h[1] = *reinterpret_cast<const v8bf*>(A + aoff + 48);

#pragma unroll
    for (int nt = 0; nt < NTC; ++nt) {
        v8f acc = {0.f, 0.f, 0.f, 0.f, 0.f, 0.f, 0.f, 0.f};
        BV16 b0, b1;
        b0.v = *reinterpret_cast<const v16bf*>(Bp + boff0 + (unsigned)nt * 512u);
        b1.v = *reinterpret_cast<const v16bf*>(Bp + boff0 + (unsigned)(NTC + nt) * 512u);
        acc = __builtin_amdgcn_wmma_f32_16x16x32_bf16(false, a0.v, false, b0.v,
                                                      (short)0, acc, false, false);
        acc = __builtin_amdgcn_wmma_f32_16x16x32_bf16(false, a1.v, false, b1.v,
                                                      (short)0, acc, false, false);
        float* crow = C + coff + (unsigned)nt * 16u;
#pragma unroll
        for (int r = 0; r < 8; ++r) crow[r * COUT] = acc[r];
    }
}

// ---------------------------------------------------------------------------
// Submanifold conv: Z[N x 64] = sum_t gather(H, nbr[:,t]) @ Wk[t]  (bf16 WMMA)
// Tap-outer loop: each gathered row is loaded exactly once (4 accumulators).
// Invalid neighbors: clamped gather (always safe) + per-lane v_cndmask zeroing
// -- no EXEC divergence, loads stay unconditional/clauseable.
// ---------------------------------------------------------------------------
__global__ void k_subm_gemm_bf16(const unsigned short* __restrict__ H,
                                 const int* __restrict__ nbr,
                                 const unsigned short* __restrict__ Wkp,
                                 float* __restrict__ Z, int N) {
    int lane = (int)(threadIdx.x & 31u);
    int wave = (int)(threadIdx.x >> 5);
    int rowbase = (blockIdx.x * (int)(blockDim.x >> 5) + wave) * 16;
    if (rowbase >= N) return;
    int lo = lane & 15, hi = lane >> 4;
    int m = rowbase + lo;

    int idx[9];
#pragma unroll
    for (int t = 0; t < 9; ++t) idx[t] = nbr[(unsigned)m * 9u + (unsigned)t];

    unsigned khalf = (unsigned)hi * 8u;
    const v8i zv = {0, 0, 0, 0, 0, 0, 0, 0};

    v8f acc0 = {0.f, 0.f, 0.f, 0.f, 0.f, 0.f, 0.f, 0.f};
    v8f acc1 = acc0, acc2 = acc0, acc3 = acc0;

#pragma unroll
    for (int t = 0; t < 9; ++t) {
        int id = idx[t];
        unsigned ao = (unsigned)(id < 0 ? 0 : id) * 64u + khalf;  // clamped: always safe
        BV16 a0, a1;
        a0.h[0] = *reinterpret_cast<const v8bf*>(H + ao);
        a0.h[1] = *reinterpret_cast<const v8bf*>(H + ao + 16);
        a1.h[0] = *reinterpret_cast<const v8bf*>(H + ao + 32);
        a1.h[1] = *reinterpret_cast<const v8bf*>(H + ao + 48);
        a0.i = (id >= 0) ? a0.i : zv;   // per-lane select, no EXEC change
        a1.i = (id >= 0) ? a1.i : zv;

        const unsigned short* wp = Wkp + (unsigned)t * 4096u + (unsigned)lane * 16u;
        BV16 b;
        b.v = *reinterpret_cast<const v16bf*>(wp);
        acc0 = __builtin_amdgcn_wmma_f32_16x16x32_bf16(false, a0.v, false, b.v, (short)0, acc0, false, false);
        b.v = *reinterpret_cast<const v16bf*>(wp + 512u);
        acc1 = __builtin_amdgcn_wmma_f32_16x16x32_bf16(false, a0.v, false, b.v, (short)0, acc1, false, false);
        b.v = *reinterpret_cast<const v16bf*>(wp + 1024u);
        acc2 = __builtin_amdgcn_wmma_f32_16x16x32_bf16(false, a0.v, false, b.v, (short)0, acc2, false, false);
        b.v = *reinterpret_cast<const v16bf*>(wp + 1536u);
        acc3 = __builtin_amdgcn_wmma_f32_16x16x32_bf16(false, a0.v, false, b.v, (short)0, acc3, false, false);
        b.v = *reinterpret_cast<const v16bf*>(wp + 2048u);
        acc0 = __builtin_amdgcn_wmma_f32_16x16x32_bf16(false, a1.v, false, b.v, (short)0, acc0, false, false);
        b.v = *reinterpret_cast<const v16bf*>(wp + 2560u);
        acc1 = __builtin_amdgcn_wmma_f32_16x16x32_bf16(false, a1.v, false, b.v, (short)0, acc1, false, false);
        b.v = *reinterpret_cast<const v16bf*>(wp + 3072u);
        acc2 = __builtin_amdgcn_wmma_f32_16x16x32_bf16(false, a1.v, false, b.v, (short)0, acc2, false, false);
        b.v = *reinterpret_cast<const v16bf*>(wp + 3584u);
        acc3 = __builtin_amdgcn_wmma_f32_16x16x32_bf16(false, a1.v, false, b.v, (short)0, acc3, false, false);
    }

    unsigned coff = (unsigned)(rowbase + hi * 8) * 64u + (unsigned)lo;
    float* crow = Z + coff;
#pragma unroll
    for (int r = 0; r < 8; ++r) {
        crow[r * 64 +  0] = acc0[r];
        crow[r * 64 + 16] = acc1[r];
        crow[r * 64 + 32] = acc2[r];
        crow[r * 64 + 48] = acc3[r];
    }
}

// ---------------------------------------------------------------------------
// BN statistics, deterministic 2-stage: per-block partial sum/sumsq per channel
// ---------------------------------------------------------------------------
__global__ void k_bn_stats(const float* __restrict__ z, float* __restrict__ partial,
                           int N, int C) {
    int tid = (int)threadIdx.x;
    int rpb = (int)blockDim.x / C;  // rows per block iteration (C in {64,256})
    int c = tid % C;
    int r0 = tid / C;
    float s = 0.f, q = 0.f;
    for (unsigned r = blockIdx.x * rpb + r0; r < (unsigned)N; r += gridDim.x * rpb) {
        float v = z[r * (unsigned)C + (unsigned)c];
        s += v; q += v * v;
    }
    __shared__ float shs[256], shq[256];
    shs[tid] = s; shq[tid] = q;
    __syncthreads();
    if (r0 == 0) {
        for (int g = 1; g < rpb; ++g) { s += shs[tid + g * C]; q += shq[tid + g * C]; }
        partial[blockIdx.x * 2u * C + (unsigned)c] = s;
        partial[blockIdx.x * 2u * C + (unsigned)(C + c)] = q;
    }
}

__global__ void k_bn_finalize(const float* __restrict__ partial, int nb,
                              const float* __restrict__ g, const float* __restrict__ b,
                              float* __restrict__ sc, float* __restrict__ sh,
                              int N, int C) {
    int c = (int)(blockIdx.x * blockDim.x + threadIdx.x);
    if (c >= C) return;
    float s = 0.f, q = 0.f;
    for (int i = 0; i < nb; ++i) {
        s += partial[(unsigned)i * 2u * C + (unsigned)c];
        q += partial[(unsigned)i * 2u * C + (unsigned)(C + c)];
    }
    float mean = s / (float)N;
    float var = q / (float)N - mean * mean;
    float scale = g[c] * rsqrtf(var + 1e-5f);
    sc[c] = scale;
    sh[c] = b[c] - mean * scale;
}

__global__ void k_bn_apply_relu_bf16(const float* __restrict__ z,
                                     const float* __restrict__ sc, const float* __restrict__ sh,
                                     unsigned short* __restrict__ h, unsigned total, int C) {
    unsigned stride = gridDim.x * blockDim.x;
    unsigned cmask = (unsigned)C - 1u;  // C is a power of two (64/256)
    for (unsigned i = blockIdx.x * blockDim.x + threadIdx.x; i < total; i += stride) {
        unsigned c = i & cmask;
        float v = z[i] * sc[c] + sh[c];
        h[i] = f2bf(v > 0.f ? v : 0.f);
    }
}

// out = relu( BN3(y) + BNs(s) ), s already resident in `out` (in-place)
__global__ void k_final_apply(const float* __restrict__ y,
                              const float* __restrict__ scA, const float* __restrict__ shA,
                              const float* __restrict__ scB, const float* __restrict__ shB,
                              float* __restrict__ out, unsigned total) {
    unsigned stride = gridDim.x * blockDim.x;
    for (unsigned i = blockIdx.x * blockDim.x + threadIdx.x; i < total; i += stride) {
        unsigned c = i & 255u;
        float v = y[i] * scA[c] + shA[c] + out[i] * scB[c] + shB[c];
        out[i] = v > 0.f ? v : 0.f;
    }
}

// ---------------------------------------------------------------------------
extern "C" void kernel_launch(void* const* d_in, const int* in_sizes, int n_in,
                              void* d_out, int out_size, void* d_ws, size_t ws_size,
                              hipStream_t stream) {
    const float* feat = (const float*)d_in[0];
    const int*   nbr  = (const int*)d_in[1];
    const float* W1 = (const float*)d_in[2];
    const float* g1 = (const float*)d_in[3];
    const float* b1 = (const float*)d_in[4];
    const float* Wk = (const float*)d_in[5];
    const float* g2 = (const float*)d_in[6];
    const float* b2 = (const float*)d_in[7];
    const float* W3 = (const float*)d_in[8];
    const float* g3 = (const float*)d_in[9];
    const float* b3 = (const float*)d_in[10];
    const float* Ws = (const float*)d_in[11];
    const float* gs = (const float*)d_in[12];
    const float* bs = (const float*)d_in[13];

    const int Cin = 64, Cmid = 64, Cout = 256;
    int N = in_sizes[0] / Cin;
    float* out = (float*)d_out;

    char* ws = (char*)d_ws;
    size_t off = 0;
    auto alloc = [&](size_t bytes) -> void* {
        void* p = ws + off;
        off = (off + bytes + 255) & ~(size_t)255;
        return p;
    };
    float*          Z    = (float*)alloc((size_t)N * Cmid * 4);  // Z1 then Z2
    float*          Y    = (float*)alloc((size_t)N * Cout * 4);
    unsigned short* Xbf  = (unsigned short*)alloc((size_t)N * Cin * 2);   // later reused as h2
    unsigned short* Hbf  = (unsigned short*)alloc((size_t)N * Cmid * 2);  // h1
    unsigned short* W1p  = (unsigned short*)alloc(2 * 4 * 512 * 2);
    unsigned short* Wkp  = (unsigned short*)alloc(9 * 2 * 4 * 512 * 2);
    unsigned short* W3p  = (unsigned short*)alloc(2 * 16 * 512 * 2);
    unsigned short* Wsp  = (unsigned short*)alloc(2 * 16 * 512 * 2);
    const int NB = 256;
    float* partial = (float*)alloc((size_t)NB * 2 * 256 * 4);
    float* scA = (float*)alloc(256 * 4);
    float* shA = (float*)alloc(256 * 4);
    float* scB = (float*)alloc(256 * 4);
    float* shB = (float*)alloc(256 * 4);

    int gemmBlocks = (N / 16 + 7) / 8;

    // 1) pack weights into WMMA B layout
    k_pack_w<<<16, 256, 0, stream>>>(W1, W1p, Cmid, 2, 4, 1);
    k_pack_w<<<64, 256, 0, stream>>>(Wk, Wkp, Cmid, 2, 4, 9);
    k_pack_w<<<64, 256, 0, stream>>>(W3, W3p, Cout, 2, 16, 1);
    k_pack_w<<<64, 256, 0, stream>>>(Ws, Wsp, Cout, 2, 16, 1);

    // 2) X -> bf16
    k_f32_to_bf16<<<2048, 256, 0, stream>>>(feat, Xbf, (unsigned)(N * Cin));

    // 3) shortcut S = X @ Ws -> d_out; stats -> (scB, shB)
    k_gemm_bf16<256, 16><<<gemmBlocks, 256, 0, stream>>>(Xbf, Wsp, out, N);
    k_bn_stats<<<NB, 256, 0, stream>>>(out, partial, N, Cout);
    k_bn_finalize<<<1, 256, 0, stream>>>(partial, NB, gs, bs, scB, shB, N, Cout);

    // 4) Z1 = X @ W1; BN+ReLU -> Hbf (bf16)
    k_gemm_bf16<64, 4><<<gemmBlocks, 256, 0, stream>>>(Xbf, W1p, Z, N);
    k_bn_stats<<<NB, 256, 0, stream>>>(Z, partial, N, Cmid);
    k_bn_finalize<<<1, 256, 0, stream>>>(partial, NB, g1, b1, scA, shA, N, Cmid);
    k_bn_apply_relu_bf16<<<2048, 256, 0, stream>>>(Z, scA, shA, Hbf, (unsigned)(N * Cmid), Cmid);

    // 5) Z2 = subm_conv(Hbf); BN+ReLU -> h2 (reuse Xbf)
    k_subm_gemm_bf16<<<gemmBlocks, 256, 0, stream>>>(Hbf, nbr, Wkp, Z, N);
    k_bn_stats<<<NB, 256, 0, stream>>>(Z, partial, N, Cmid);
    k_bn_finalize<<<1, 256, 0, stream>>>(partial, NB, g2, b2, scA, shA, N, Cmid);
    k_bn_apply_relu_bf16<<<2048, 256, 0, stream>>>(Z, scA, shA, Xbf, (unsigned)(N * Cmid), Cmid);

    // 6) Y = h2 @ W3; stats -> (scA, shA)
    k_gemm_bf16<256, 16><<<gemmBlocks, 256, 0, stream>>>(Xbf, W3p, Y, N);
    k_bn_stats<<<NB, 256, 0, stream>>>(Y, partial, N, Cout);
    k_bn_finalize<<<1, 256, 0, stream>>>(partial, NB, g3, b3, scA, shA, N, Cout);

    // 7) out = relu(BN(Y) + BN(S))  (S in-place in d_out)
    k_final_apply<<<2048, 256, 0, stream>>>(Y, scA, shA, scB, shB, out, (unsigned)(N * Cout));
}
